// GraphAttentionLayer_57698590654935
// MI455X (gfx1250) — matrix operand order repaired and checked by reading, not compiled, not run on hardware
//
#include <hip/hip_runtime.h>

typedef __attribute__((ext_vector_type(16))) __bf16 v16bf;
typedef __attribute__((ext_vector_type(8)))  __bf16 bf16x8;
typedef __attribute__((ext_vector_type(8)))  float  v8f;
typedef __attribute__((ext_vector_type(4)))  unsigned int u32x4;
typedef __attribute__((ext_vector_type(8)))  int i32x8;
typedef __attribute__((ext_vector_type(4)))  int i32x4;

#define NROW 8192
#define INF  512
#define OUTF 256
#define NEGBIG (-9.0e15f)

// workspace layout (bytes)
#define HT_OFF 0                      // bf16 h_t[256][8192]  = 4,194,304 B
#define WT_OFF 4194304                // bf16 W_t[256][512]   =   262,144 B
#define E1_OFF 4456448                // f32  e1[8192]        =    32,768 B
#define E2_OFF 4489216                // f32  e2[8192]        =    32,768 B

#if __has_builtin(__builtin_amdgcn_tensor_load_to_lds) && __has_builtin(__builtin_amdgcn_s_wait_tensorcnt)
#define HAVE_TDM 1
#else
#define HAVE_TDM 0
#endif

// ---------------- kernel 0: W (512x256 f32) -> W_t (256x512 bf16) ----------
__global__ void k_prepW(const float* __restrict__ W, __bf16* __restrict__ Wt) {
    int idx = blockIdx.x * 256 + threadIdx.x;   // 131072 elements
    int n = idx >> 9;          // 0..255
    int k = idx & 511;         // 0..511
    Wt[(size_t)n * INF + k] = (__bf16)W[(size_t)k * OUTF + n];
}

// ---------------- kernel 1: h = x @ W, write h_t (bf16, transposed) --------
__global__ __launch_bounds__(32) void k_hgemm(const float* __restrict__ x,
                                              const __bf16* __restrict__ Wt,
                                              __bf16* __restrict__ ht) {
    const int lane = threadIdx.x & 31;
    const int l16  = lane & 15;
    const int half = lane >> 4;
    const int rowbase = blockIdx.x * 16;

    const float* xrow = x + (size_t)(rowbase + l16) * INF;

    v8f acc[16];
    const v8f zero = {0.f,0.f,0.f,0.f,0.f,0.f,0.f,0.f};
#pragma unroll
    for (int t = 0; t < 16; ++t) acc[t] = zero;

    for (int kb = 0; kb < INF; kb += 32) {
        const int ka = kb + 16 * half;          // A-fragment K ordering
        float4 x0 = *(const float4*)(xrow + ka);
        float4 x1 = *(const float4*)(xrow + ka + 4);
        float4 x2 = *(const float4*)(xrow + ka + 8);
        float4 x3 = *(const float4*)(xrow + ka + 12);
        v16bf a;
        a[0]=(__bf16)x0.x;  a[1]=(__bf16)x0.y;  a[2]=(__bf16)x0.z;  a[3]=(__bf16)x0.w;
        a[4]=(__bf16)x1.x;  a[5]=(__bf16)x1.y;  a[6]=(__bf16)x1.z;  a[7]=(__bf16)x1.w;
        a[8]=(__bf16)x2.x;  a[9]=(__bf16)x2.y;  a[10]=(__bf16)x2.z; a[11]=(__bf16)x2.w;
        a[12]=(__bf16)x3.x; a[13]=(__bf16)x3.y; a[14]=(__bf16)x3.z; a[15]=(__bf16)x3.w;

        const int kbB = kb + 16 * half;         // B-fragment K ordering
#pragma unroll
        for (int t = 0; t < 16; ++t) {
            v16bf b = *(const v16bf*)(Wt + (size_t)(t * 16 + l16) * INF + kbB);
            acc[t] = __builtin_amdgcn_wmma_f32_16x16x32_bf16(
                false, a, false, b, (short)0, acc[t], false, false);
        }
    }

#pragma unroll
    for (int t = 0; t < 16; ++t) {
        bf16x8 st;
#pragma unroll
        for (int r = 0; r < 8; ++r) st[r] = (__bf16)acc[t][r];
        *(bf16x8*)(ht + (size_t)(t * 16 + l16) * NROW + rowbase + 8 * half) = st;
    }
}

// ---------------- kernel 2: e1 = h@a1, e2 = h@a2 ---------------------------
__global__ void k_evec(const __bf16* __restrict__ ht, const float* __restrict__ a,
                       float* __restrict__ e1, float* __restrict__ e2) {
    int i = blockIdx.x * 256 + threadIdx.x;     // 8192 rows
    float s1 = 0.f, s2 = 0.f;
    for (int n = 0; n < OUTF; ++n) {
        float hv = (float)ht[(size_t)n * NROW + i];   // coalesced over i
        s1 += hv * a[n];
        s2 += hv * a[OUTF + n];
    }
    e1[i] = s1;
    e2[i] = s2;
}

#if HAVE_TDM
// Issue one 2-D TDM load: 256 rows x 64 B (16 dwords) of h_t, row stride 16 KB,
// packed contiguously into a 16 KB LDS slab.  D# layout per ISA ch.8.
__device__ __forceinline__ void tdm_issue(unsigned lds_addr, const void* gptr) {
    unsigned long long ga = (unsigned long long)gptr;
    u32x4 g0;
    g0[0] = 1u;                                        // count=1 (valid user D#)
    g0[1] = lds_addr;                                  // lds_addr [63:32]
    g0[2] = (unsigned)ga;                              // global_addr [95:64]
    g0[3] = ((unsigned)(ga >> 32) & 0x01FFFFFFu)       // global_addr [120:96]
            | 0x80000000u;                             // type=2 ("image") [127:126]
    i32x8 g1;
    g1[0] = 0x00020000;   // workgroup_mask=0, data_size=2 (4-byte units)
    g1[1] = 0x10000000;   // tensor_dim0 = 4096 dwords  -> bits[63:48]
    g1[2] = 0x01000000;   // tensor_dim1 = 256          -> bits[95:80]
    g1[3] = 0x00100000;   // tile_dim0   = 16 dwords    -> bits[127:112]
    g1[4] = 0x00000100;   // tile_dim1   = 256          -> bits[143:128]
    g1[5] = 4096;         // tensor_dim0_stride = 4096 dwords (16 KB row pitch)
    g1[6] = 0;
    g1[7] = 0;
    i32x4 z4 = {0, 0, 0, 0};
    i32x8 z8 = {0, 0, 0, 0, 0, 0, 0, 0};
    __builtin_amdgcn_tensor_load_to_lds(g0, g1, z4, z4, z8, 0);
}
#endif

// ---------------- kernel 3: h' = where(adj>0, e1[i]+e2[j], -9e15) @ h ------
// 512 blocks (one per 16-row tile), 4 waves: K split 4x2048.
// B slab (32 K x 256 N bf16 = 16 KB) is TDM-DMA'd into per-wave double-buffered
// LDS, overlapped with A-fragment construction via TENSORcnt.
__global__ __launch_bounds__(128) void k_att(const int* __restrict__ adj,
                                             const __bf16* __restrict__ ht,
                                             const float* __restrict__ e1,
                                             const float* __restrict__ e2,
                                             float* __restrict__ out) {
    extern __shared__ char ldsbuf[];            // 128 KB: 4 waves x 2 bufs x 16 KB
    const int tid  = threadIdx.x;
    const int wave = tid >> 5;
    const int lane = tid & 31;
    const int l16  = lane & 15;
    const int half = lane >> 4;
    const int row  = blockIdx.x * 16 + l16;

    const int* adjrow = adj + (size_t)row * NROW;
    const float e1v   = e1[row];

    v8f acc[16];
    const v8f zero = {0.f,0.f,0.f,0.f,0.f,0.f,0.f,0.f};
#pragma unroll
    for (int t = 0; t < 16; ++t) acc[t] = zero;

    const int k0 = wave * 2048, kend = k0 + 2048;

#if HAVE_TDM
    unsigned ldsb;
    {
        __attribute__((address_space(3))) char* p3 =
            (__attribute__((address_space(3))) char*)ldsbuf;
        ldsb = (unsigned)(unsigned long long)p3;
    }
    const unsigned woff = ldsb + (unsigned)wave * 32768u;
    const char* wslab = ldsbuf + wave * 32768;
    tdm_issue(woff, (const char*)ht + (size_t)k0 * 2);   // prime buffer 0
#endif

    for (int kb = k0; kb < kend; kb += 32) {
        const int ka = kb + 16 * half;
        __builtin_prefetch(adjrow + ka + 512, 0, 1);     // adj is the HBM stream
        int4   i0 = *(const int4*)(adjrow + ka);
        int4   i1 = *(const int4*)(adjrow + ka + 4);
        int4   i2 = *(const int4*)(adjrow + ka + 8);
        int4   i3 = *(const int4*)(adjrow + ka + 12);
        float4 f0 = *(const float4*)(e2 + ka);
        float4 f1 = *(const float4*)(e2 + ka + 4);
        float4 f2 = *(const float4*)(e2 + ka + 8);
        float4 f3 = *(const float4*)(e2 + ka + 12);

        v16bf a;
        a[0]  = (__bf16)(i0.x > 0 ? e1v + f0.x : NEGBIG);
        a[1]  = (__bf16)(i0.y > 0 ? e1v + f0.y : NEGBIG);
        a[2]  = (__bf16)(i0.z > 0 ? e1v + f0.z : NEGBIG);
        a[3]  = (__bf16)(i0.w > 0 ? e1v + f0.w : NEGBIG);
        a[4]  = (__bf16)(i1.x > 0 ? e1v + f1.x : NEGBIG);
        a[5]  = (__bf16)(i1.y > 0 ? e1v + f1.y : NEGBIG);
        a[6]  = (__bf16)(i1.z > 0 ? e1v + f1.z : NEGBIG);
        a[7]  = (__bf16)(i1.w > 0 ? e1v + f1.w : NEGBIG);
        a[8]  = (__bf16)(i2.x > 0 ? e1v + f2.x : NEGBIG);
        a[9]  = (__bf16)(i2.y > 0 ? e1v + f2.y : NEGBIG);
        a[10] = (__bf16)(i2.z > 0 ? e1v + f2.z : NEGBIG);
        a[11] = (__bf16)(i2.w > 0 ? e1v + f2.w : NEGBIG);
        a[12] = (__bf16)(i3.x > 0 ? e1v + f3.x : NEGBIG);
        a[13] = (__bf16)(i3.y > 0 ? e1v + f3.y : NEGBIG);
        a[14] = (__bf16)(i3.z > 0 ? e1v + f3.z : NEGBIG);
        a[15] = (__bf16)(i3.w > 0 ? e1v + f3.w : NEGBIG);

#if HAVE_TDM
        const int bufc = (kb >> 5) & 1;
        if (kb + 32 < kend) {                            // prefetch next slab
            tdm_issue(woff + (unsigned)((bufc ^ 1) * 16384),
                      (const char*)ht + (size_t)(kb + 32) * 2);
            __builtin_amdgcn_s_wait_tensorcnt(1);        // current slab ready
        } else {
            __builtin_amdgcn_s_wait_tensorcnt(0);
        }
        const char* slab = wslab + bufc * 16384;
#pragma unroll
        for (int t = 0; t < 16; ++t) {
            // row (t*16+l16): 64 B of K; halves at +0 / +32 B
            v16bf b = *(const v16bf*)(slab + (t * 16 + l16) * 64 + 32 * half);
            acc[t] = __builtin_amdgcn_wmma_f32_16x16x32_bf16(
                false, a, false, b, (short)0, acc[t], false, false);
        }
#else
        const int kbB = kb + 16 * half;
#pragma unroll
        for (int t = 0; t < 16; ++t) {
            v16bf b = *(const v16bf*)(ht + (size_t)(t * 16 + l16) * NROW + kbB);
            acc[t] = __builtin_amdgcn_wmma_f32_16x16x32_bf16(
                false, a, false, b, (short)0, acc[t], false, false);
        }
#endif
    }

#if HAVE_TDM
    __builtin_amdgcn_s_wait_tensorcnt(0);  // LDS quiescent before reuse
#endif
    __syncthreads();

    // deterministic phased reduction across the 4 K-split waves (reuses LDS)
    float* red = (float*)ldsbuf;            // 16 KB
    for (int w = 0; w < 4; ++w) {
        if (wave == w) {
#pragma unroll
            for (int t = 0; t < 16; ++t) {
#pragma unroll
                for (int r = 0; r < 8; ++r) {
                    int off = (8 * half + r) * OUTF + t * 16 + l16;
                    if (w == 0) red[off] = acc[t][r];
                    else        red[off] += acc[t][r];
                }
            }
        }
        __syncthreads();
    }

    float* outrow = out + (size_t)blockIdx.x * 16 * OUTF;
    for (int idx = tid; idx < 16 * OUTF; idx += 128) outrow[idx] = red[idx];
}

// ---------------- launcher --------------------------------------------------
extern "C" void kernel_launch(void* const* d_in, const int* in_sizes, int n_in,
                              void* d_out, int out_size, void* d_ws, size_t ws_size,
                              hipStream_t stream) {
    const float* x   = (const float*)d_in[0];
    const float* W   = (const float*)d_in[1];
    const float* a   = (const float*)d_in[2];
    const int*   adj = (const int*)d_in[3];
    float*       out = (float*)d_out;

    char* ws = (char*)d_ws;
    __bf16* ht = (__bf16*)(ws + HT_OFF);
    __bf16* Wt = (__bf16*)(ws + WT_OFF);
    float*  e1 = (float*)(ws + E1_OFF);
    float*  e2 = (float*)(ws + E2_OFF);

    k_prepW<<<512, 256, 0, stream>>>(W, Wt);
    k_hgemm<<<512, 32, 0, stream>>>(x, Wt, ht);
    k_evec <<<32, 256, 0, stream>>>(ht, a, e1, e2);
    k_att  <<<512, 128, 131072, stream>>>(adj, ht, e1, e2, out);
}
